// RNNStack_63299228009388
// MI455X (gfx1250) — matrix operand order repaired and checked by reading, not compiled, not run on hardware
//
#include <hip/hip_runtime.h>

// ---------------------------------------------------------------------------
// CDNA5 (gfx1250) GRU stack: bf16 WMMA GEMMs + fp32 fused GRU epilogues.
// Round 3:
//  * Fused step launches: K_t = { gru1(t), gru0(t+1) } (independent halves,
//    dispatched on blockIdx.z) -> sequential chain 1024 -> 513 kernels,
//    2x workgroups per launch.
//  * mx0_gemm register-blocks N by 2 (8 accumulators) to halve A-operand
//    re-reads and raise WMMA:load ratio.
//  * All inner loops remain software-pipelined (register double-buffered).
// ---------------------------------------------------------------------------

#define DEVFN __device__ __forceinline__

typedef __bf16 bf16_t;
typedef __attribute__((ext_vector_type(16))) __bf16 bf16x16;
typedef __attribute__((ext_vector_type(8)))  __bf16 bf16x8;
typedef __attribute__((ext_vector_type(8)))  float  f32x8;

constexpr int B_  = 64;
constexpr int T_  = 512;
constexpr int D_  = 1024;
constexpr int H_  = 1024;
constexpr int H3_ = 3072;

DEVFN f32x8 wmma_bf16(bf16x16 a, bf16x16 b, f32x8 c) {
  // 8 args: (neg_a, A, neg_b, B, c_mod, C, reuse_a, reuse_b)
  return __builtin_amdgcn_wmma_f32_16x16x32_bf16(false, a, false, b, (short)0, c,
                                                 false, false);
}

// A-operand fragment (16x32 bf16). prow points at this lane's row, K-base.
DEVFN bf16x16 load_a_frag(const bf16_t* prow, int half) {
  bf16x8 lo = *(const bf16x8*)(prow + 8 * half);
  bf16x8 hi = *(const bf16x8*)(prow + 16 + 8 * half);
  bf16x16 f;
#pragma unroll
  for (int i = 0; i < 8; ++i) { f[i] = lo[i]; f[i + 8] = hi[i]; }
  return f;
}

// B-operand fragment (32x16 bf16) from N-major weights; K = 8*half + e (e<8), 16 + 8*half + (e-8) (e>=8).
DEVFN bf16x16 load_b_frag(const bf16_t* pcol, int half) {
  bf16x8 lo = *(const bf16x8*)(pcol + 8 * half);
  bf16x8 hi = *(const bf16x8*)(pcol + 16 + 8 * half);
  bf16x16 f;
#pragma unroll
  for (int i = 0; i < 8; ++i) { f[i] = lo[i]; f[i + 8] = hi[i]; }
  return f;
}

DEVFN float sigmoidf_(float x) { return 1.0f / (1.0f + __expf(-x)); }

// ---------------------------------------------------------------------------
// Setup kernels
// ---------------------------------------------------------------------------
__global__ void convert_to_bf16(const float* __restrict__ s,
                                bf16_t* __restrict__ d, int n) {
  int i = blockIdx.x * blockDim.x + threadIdx.x;
  if (i < n) d[i] = (bf16_t)s[i];
}

// src is (K,N) row-major fp32 -> dst is (N,K) row-major bf16
__global__ void transpose_to_bf16(const float* __restrict__ s,
                                  bf16_t* __restrict__ d, int K, int N) {
  int i = blockIdx.x * blockDim.x + threadIdx.x;
  if (i < K * N) {
    int k = i / N, n = i % N;
    d[(size_t)n * K + k] = (bf16_t)s[i];
  }
}

__global__ void zero_state(float* h0f, float* h1f, bf16_t* a, bf16_t* b,
                           bf16_t* c, bf16_t* d, int n) {
  int i = blockIdx.x * blockDim.x + threadIdx.x;
  if (i < n) {
    h0f[i] = 0.f; h1f[i] = 0.f;
    bf16_t z = (bf16_t)0.f;
    a[i] = z; b[i] = z; c[i] = z; d[i] = z;
  }
}

// ---------------------------------------------------------------------------
// Bulk GEMM: mx0[t][b][:] = x[b][t][:] @ k0 + b0[0]
//   xbf : (B*T, D) bf16    k0T : (3H, D) bf16 (N-major)    mx0 : (T, B, 3H)
// Each wave: 64 rows x 32 cols (8 accumulators; 4 A frags x 2 B frags).
// grid (H3/32, B*T/256), block 128. Software-pipelined.
// ---------------------------------------------------------------------------
__global__ __launch_bounds__(128) void mx0_gemm(const bf16_t* __restrict__ xbf,
                                                const bf16_t* __restrict__ kT,
                                                const float* __restrict__ bias,
                                                float* __restrict__ mx0) {
  const int tid = threadIdx.x, lane = tid & 31, wave = tid >> 5, half = lane >> 4;
  const int colsel = lane & 15;
  const int j = blockIdx.x * 32;
  const int m0 = (blockIdx.y * 4 + wave) * 64;

  const bf16_t* bc0 = kT + (size_t)(j + colsel) * D_;
  const bf16_t* bc1 = bc0 + (size_t)16 * D_;
  const bf16_t* ap0 = xbf + (size_t)(m0 + colsel) * D_;

  f32x8 z8 = {0.f,0.f,0.f,0.f,0.f,0.f,0.f,0.f};
  f32x8 ac00 = z8, ac10 = z8, ac20 = z8, ac30 = z8;
  f32x8 ac01 = z8, ac11 = z8, ac21 = z8, ac31 = z8;

  // pipeline prologue
  bf16x16 b0C = load_b_frag(bc0, half);
  bf16x16 b1C = load_b_frag(bc1, half);
  bf16x16 a0C = load_a_frag(ap0, half);
  bf16x16 a1C = load_a_frag(ap0 + (size_t)16 * D_, half);
  bf16x16 a2C = load_a_frag(ap0 + (size_t)32 * D_, half);
  bf16x16 a3C = load_a_frag(ap0 + (size_t)48 * D_, half);
#pragma unroll 4
  for (int kk = 0; kk < D_; kk += 32) {
    const int kn = (kk + 32 < D_) ? kk + 32 : 0;  // wrap on last iter (no OOB)
    bf16x16 b0N = load_b_frag(bc0 + kn, half);
    bf16x16 b1N = load_b_frag(bc1 + kn, half);
    bf16x16 a0N = load_a_frag(ap0 + kn, half);
    bf16x16 a1N = load_a_frag(ap0 + (size_t)16 * D_ + kn, half);
    bf16x16 a2N = load_a_frag(ap0 + (size_t)32 * D_ + kn, half);
    bf16x16 a3N = load_a_frag(ap0 + (size_t)48 * D_ + kn, half);
    ac00 = wmma_bf16(a0C, b0C, ac00);
    ac10 = wmma_bf16(a1C, b0C, ac10);
    ac20 = wmma_bf16(a2C, b0C, ac20);
    ac30 = wmma_bf16(a3C, b0C, ac30);
    ac01 = wmma_bf16(a0C, b1C, ac01);
    ac11 = wmma_bf16(a1C, b1C, ac11);
    ac21 = wmma_bf16(a2C, b1C, ac21);
    ac31 = wmma_bf16(a3C, b1C, ac31);
    b0C = b0N; b1C = b1N; a0C = a0N; a1C = a1N; a2C = a2N; a3C = a3N;
  }
  f32x8 acc[2][4] = {{ac00, ac10, ac20, ac30}, {ac01, ac11, ac21, ac31}};
#pragma unroll
  for (int nt = 0; nt < 2; ++nt) {
    const int cg = j + nt * 16 + colsel;
    const float bv = bias[cg];
#pragma unroll
    for (int tl = 0; tl < 4; ++tl) {
#pragma unroll
      for (int i = 0; i < 8; ++i) {
        int m  = m0 + tl * 16 + i + 8 * half;  // row = b*T + t
        int bb = m >> 9;                       // /T (T=512)
        int tt = m & (T_ - 1);
        mx0[((size_t)tt * B_ + bb) * H3_ + cg] = acc[nt][tl][i] + bv;
      }
    }
  }
}

// ---------------------------------------------------------------------------
// GRU step bodies (inlined; shared by standalone + fused kernels).
// 4 waves split K (256 each); LDS partial reduction; fused gate epilogue.
// Block mapping: blockIdx.x = H column slice (j/16), blockIdx.y = batch/16.
// ---------------------------------------------------------------------------
DEVFN void gru0_body(float* sm,                  // >= 4*3*16*16 floats
                     const float* __restrict__ mx_t,
                     const bf16_t* __restrict__ rkT,
                     const float* __restrict__ bias,
                     const int* __restrict__ len,
                     const bf16_t* __restrict__ hin_bf,
                     float* __restrict__ hf,
                     bf16_t* __restrict__ hout_bf, int t) {
  constexpr int NG = 3;
  const int tid = threadIdx.x, lane = tid & 31, wave = tid >> 5, half = lane >> 4;
  const int colsel = lane & 15;
  const int j = blockIdx.x * 16;
  const int mrow = blockIdx.y * 16;
  constexpr int KQ = H_ / 4;
  const int kbase = wave * KQ;

  const bf16_t* ap = hin_bf + (size_t)(mrow + colsel) * H_ + kbase;
  const bf16_t* bz = rkT + (size_t)(0 * H_ + j + colsel) * H_ + kbase;
  const bf16_t* br = rkT + (size_t)(1 * H_ + j + colsel) * H_ + kbase;
  const bf16_t* bh = rkT + (size_t)(2 * H_ + j + colsel) * H_ + kbase;

  f32x8 z8 = {0.f,0.f,0.f,0.f,0.f,0.f,0.f,0.f};
  f32x8 az = z8, ar = z8, ah = z8;

  bf16x16 aC  = load_a_frag(ap, half);
  bf16x16 bzC = load_b_frag(bz, half);
  bf16x16 brC = load_b_frag(br, half);
  bf16x16 bhC = load_b_frag(bh, half);
#pragma unroll
  for (int kk = 0; kk < KQ; kk += 32) {
    const int kn = (kk + 32 < KQ) ? kk + 32 : 0;
    bf16x16 aN  = load_a_frag(ap + kn, half);
    bf16x16 bzN = load_b_frag(bz + kn, half);
    bf16x16 brN = load_b_frag(br + kn, half);
    bf16x16 bhN = load_b_frag(bh + kn, half);
    az = wmma_bf16(aC, bzC, az);
    ar = wmma_bf16(aC, brC, ar);
    ah = wmma_bf16(aC, bhC, ah);
    aC = aN; bzC = bzN; brC = brN; bhC = bhN;
  }
#pragma unroll
  for (int i = 0; i < 8; ++i) {
    int r = i + 8 * half;
    sm[(((wave * NG + 0) * 16) + r) * 16 + colsel] = az[i];
    sm[(((wave * NG + 1) * 16) + r) * 16 + colsel] = ar[i];
    sm[(((wave * NG + 2) * 16) + r) * 16 + colsel] = ah[i];
  }
  __syncthreads();
#pragma unroll
  for (int e = tid; e < 256; e += 128) {
    int r = e >> 4, c = e & 15;
    float s0 = 0.f, s1 = 0.f, s2 = 0.f;
#pragma unroll
    for (int w = 0; w < 4; ++w) {
      s0 += sm[(((w * NG + 0) * 16) + r) * 16 + c];
      s1 += sm[(((w * NG + 1) * 16) + r) * 16 + c];
      s2 += sm[(((w * NG + 2) * 16) + r) * 16 + c];
    }
    int b = mrow + r;
    int cg = j + c;
    const float* mx = mx_t + (size_t)b * H3_;
    float z    = sigmoidf_(mx[cg] + s0 + bias[H3_ + cg]);
    float rg   = sigmoidf_(mx[H_ + cg] + s1 + bias[H3_ + H_ + cg]);
    float cand = tanhf(mx[2 * H_ + cg] + rg * (s2 + bias[H3_ + 2 * H_ + cg]));
    float hold = hf[(size_t)b * H_ + cg];
    float hnew = z * hold + (1.0f - z) * cand;
    float hn   = (t < len[b]) ? hnew : hold;
    hf[(size_t)b * H_ + cg]      = hn;
    hout_bf[(size_t)b * H_ + cg] = (bf16_t)hn;
  }
}

DEVFN void gru1_body(float* sm,                  // >= 4*6*16*16 floats
                     const bf16_t* __restrict__ x1bf,
                     const bf16_t* __restrict__ h1in,
                     const bf16_t* __restrict__ kT,
                     const bf16_t* __restrict__ rkT,
                     const float* __restrict__ bias,
                     const int* __restrict__ len,
                     float* __restrict__ hf,
                     bf16_t* __restrict__ hout_bf,
                     float* __restrict__ out, int t) {
  constexpr int NG = 6;
  const int tid = threadIdx.x, lane = tid & 31, wave = tid >> 5, half = lane >> 4;
  const int colsel = lane & 15;
  const int j = blockIdx.x * 16;
  const int mrow = blockIdx.y * 16;
  constexpr int KQ = H_ / 4;
  const int kbase = wave * KQ;

  const bf16_t* xp  = x1bf + (size_t)(mrow + colsel) * H_ + kbase;
  const bf16_t* hp  = h1in + (size_t)(mrow + colsel) * H_ + kbase;
  const bf16_t* kz  = kT  + (size_t)(0 * H_ + j + colsel) * H_ + kbase;
  const bf16_t* kr  = kT  + (size_t)(1 * H_ + j + colsel) * H_ + kbase;
  const bf16_t* kh  = kT  + (size_t)(2 * H_ + j + colsel) * H_ + kbase;
  const bf16_t* rz  = rkT + (size_t)(0 * H_ + j + colsel) * H_ + kbase;
  const bf16_t* rr_ = rkT + (size_t)(1 * H_ + j + colsel) * H_ + kbase;
  const bf16_t* rh  = rkT + (size_t)(2 * H_ + j + colsel) * H_ + kbase;

  f32x8 z8 = {0.f,0.f,0.f,0.f,0.f,0.f,0.f,0.f};
  f32x8 axz = z8, axr = z8, axh = z8, ahz = z8, ahr = z8, ahh = z8;

  bf16x16 a0C = load_a_frag(xp, half);
  bf16x16 a1C = load_a_frag(hp, half);
  bf16x16 kzC = load_b_frag(kz, half);
  bf16x16 krC = load_b_frag(kr, half);
  bf16x16 khC = load_b_frag(kh, half);
  bf16x16 rzC = load_b_frag(rz, half);
  bf16x16 rrC = load_b_frag(rr_, half);
  bf16x16 rhC = load_b_frag(rh, half);
#pragma unroll
  for (int kk = 0; kk < KQ; kk += 32) {
    const int kn = (kk + 32 < KQ) ? kk + 32 : 0;
    bf16x16 a0N = load_a_frag(xp + kn, half);
    bf16x16 a1N = load_a_frag(hp + kn, half);
    bf16x16 kzN = load_b_frag(kz + kn, half);
    bf16x16 krN = load_b_frag(kr + kn, half);
    bf16x16 khN = load_b_frag(kh + kn, half);
    bf16x16 rzN = load_b_frag(rz + kn, half);
    bf16x16 rrN = load_b_frag(rr_ + kn, half);
    bf16x16 rhN = load_b_frag(rh + kn, half);
    axz = wmma_bf16(a0C, kzC, axz);
    axr = wmma_bf16(a0C, krC, axr);
    axh = wmma_bf16(a0C, khC, axh);
    ahz = wmma_bf16(a1C, rzC, ahz);
    ahr = wmma_bf16(a1C, rrC, ahr);
    ahh = wmma_bf16(a1C, rhC, ahh);
    a0C = a0N; a1C = a1N;
    kzC = kzN; krC = krN; khC = khN;
    rzC = rzN; rrC = rrN; rhC = rhN;
  }
#pragma unroll
  for (int i = 0; i < 8; ++i) {
    int r = i + 8 * half;
    sm[(((wave * NG + 0) * 16) + r) * 16 + colsel] = axz[i];
    sm[(((wave * NG + 1) * 16) + r) * 16 + colsel] = axr[i];
    sm[(((wave * NG + 2) * 16) + r) * 16 + colsel] = axh[i];
    sm[(((wave * NG + 3) * 16) + r) * 16 + colsel] = ahz[i];
    sm[(((wave * NG + 4) * 16) + r) * 16 + colsel] = ahr[i];
    sm[(((wave * NG + 5) * 16) + r) * 16 + colsel] = ahh[i];
  }
  __syncthreads();
#pragma unroll
  for (int e = tid; e < 256; e += 128) {
    int r = e >> 4, c = e & 15;
    float s[6] = {0.f, 0.f, 0.f, 0.f, 0.f, 0.f};
#pragma unroll
    for (int w = 0; w < 4; ++w)
#pragma unroll
      for (int g = 0; g < 6; ++g)
        s[g] += sm[(((w * NG + g) * 16) + r) * 16 + c];
    int b = mrow + r;
    int cg = j + c;
    float z    = sigmoidf_(s[0] + bias[cg] + s[3] + bias[H3_ + cg]);
    float rg   = sigmoidf_(s[1] + bias[H_ + cg] + s[4] + bias[H3_ + H_ + cg]);
    float cand = tanhf(s[2] + bias[2 * H_ + cg] + rg * (s[5] + bias[H3_ + 2 * H_ + cg]));
    float hold = hf[(size_t)b * H_ + cg];
    float hnew = z * hold + (1.0f - z) * cand;
    float hn   = (t < len[b]) ? hnew : hold;
    hf[(size_t)b * H_ + cg]             = hn;
    hout_bf[(size_t)b * H_ + cg]        = (bf16_t)hn;
    out[((size_t)b * T_ + t) * H_ + cg] = hn;
  }
}

// Standalone wrappers (scan prologue / epilogue).
__global__ __launch_bounds__(128) void gru0_step(
    const float* __restrict__ mx_t, const bf16_t* __restrict__ rkT,
    const float* __restrict__ bias, const int* __restrict__ len,
    const bf16_t* __restrict__ hin_bf, float* __restrict__ hf,
    bf16_t* __restrict__ hout_bf, int t) {
  __shared__ float sm[4 * 3 * 16 * 16];
  gru0_body(sm, mx_t, rkT, bias, len, hin_bf, hf, hout_bf, t);
}

__global__ __launch_bounds__(128) void gru1_step(
    const bf16_t* __restrict__ x1bf, const bf16_t* __restrict__ h1in,
    const bf16_t* __restrict__ kT, const bf16_t* __restrict__ rkT,
    const float* __restrict__ bias, const int* __restrict__ len,
    float* __restrict__ hf, bf16_t* __restrict__ hout_bf,
    float* __restrict__ out, int t) {
  __shared__ float sm[4 * 6 * 16 * 16];
  gru1_body(sm, x1bf, h1in, kT, rkT, bias, len, hf, hout_bf, out, t);
}

// Fused step: blockIdx.z == 0 -> gru1(t); blockIdx.z == 1 -> gru0(t+1).
// Both halves depend only on outputs of the previous launch:
//   gru1(t):  reads h0cur (= h0 after t), h1cur;  writes h1nxt, out[t]
//   gru0(t+1): reads h0cur;                      writes h0nxt (other buffer)
__global__ __launch_bounds__(128) void gru_fused(
    const float* __restrict__ mx_next,   // mx0 slice for step t+1
    const bf16_t* __restrict__ rk0T, const float* __restrict__ bias0,
    const bf16_t* __restrict__ k1T, const bf16_t* __restrict__ rk1T,
    const float* __restrict__ bias1, const int* __restrict__ len,
    const bf16_t* __restrict__ h0cur, bf16_t* __restrict__ h0nxt,
    float* __restrict__ h0f,
    const bf16_t* __restrict__ h1cur, bf16_t* __restrict__ h1nxt,
    float* __restrict__ h1f,
    float* __restrict__ out, int t) {
  __shared__ float sm[4 * 6 * 16 * 16];
  if (blockIdx.z == 0) {
    gru1_body(sm, h0cur, h1cur, k1T, rk1T, bias1, len, h1f, h1nxt, out, t);
  } else {
    gru0_body(sm, mx_next, rk0T, bias0, len, h0cur, h0f, h0nxt, t + 1);
  }
}

// ---------------------------------------------------------------------------
// Host driver
// ---------------------------------------------------------------------------
extern "C" void kernel_launch(void* const* d_in, const int* in_sizes, int n_in,
                              void* d_out, int out_size, void* d_ws,
                              size_t ws_size, hipStream_t stream) {
  (void)in_sizes; (void)n_in; (void)out_size; (void)ws_size;

  const float* x   = (const float*)d_in[0];
  const int*   len = (const int*)d_in[1];
  const float* k0  = (const float*)d_in[2];
  const float* rk0 = (const float*)d_in[3];
  const float* b0  = (const float*)d_in[4];
  const float* k1  = (const float*)d_in[5];
  const float* rk1 = (const float*)d_in[6];
  const float* b1  = (const float*)d_in[7];
  float* out = (float*)d_out;

  char* p = (char*)d_ws;
  auto take = [&](size_t bytes) {
    char* r = p;
    p += (bytes + 255) & ~(size_t)255;
    return r;
  };
  bf16_t* xbf  = (bf16_t*)take((size_t)B_ * T_ * D_ * 2);
  bf16_t* k0T  = (bf16_t*)take((size_t)H3_ * D_ * 2);
  bf16_t* rk0T = (bf16_t*)take((size_t)H3_ * H_ * 2);
  bf16_t* k1T  = (bf16_t*)take((size_t)H3_ * H_ * 2);
  bf16_t* rk1T = (bf16_t*)take((size_t)H3_ * H_ * 2);
  float*  mx0  = (float*)take((size_t)T_ * B_ * H3_ * 4);
  float*  h0f  = (float*)take((size_t)B_ * H_ * 4);
  float*  h1f  = (float*)take((size_t)B_ * H_ * 4);
  bf16_t* h0bf[2] = {(bf16_t*)take((size_t)B_ * H_ * 2),
                     (bf16_t*)take((size_t)B_ * H_ * 2)};
  bf16_t* h1bf[2] = {(bf16_t*)take((size_t)B_ * H_ * 2),
                     (bf16_t*)take((size_t)B_ * H_ * 2)};

  {
    int n = B_ * T_ * D_;
    convert_to_bf16<<<(n + 255) / 256, 256, 0, stream>>>(x, xbf, n);
  }
  transpose_to_bf16<<<(D_ * H3_ + 255) / 256, 256, 0, stream>>>(k0, k0T, D_, H3_);
  transpose_to_bf16<<<(H_ * H3_ + 255) / 256, 256, 0, stream>>>(rk0, rk0T, H_, H3_);
  transpose_to_bf16<<<(H_ * H3_ + 255) / 256, 256, 0, stream>>>(k1, k1T, H_, H3_);
  transpose_to_bf16<<<(H_ * H3_ + 255) / 256, 256, 0, stream>>>(rk1, rk1T, H_, H3_);
  zero_state<<<(B_ * H_ + 255) / 256, 256, 0, stream>>>(h0f, h1f, h0bf[0],
                                                        h0bf[1], h1bf[0], h1bf[1],
                                                        B_ * H_);

  // Bulk input projection for layer 0 (no sequential dependency).
  mx0_gemm<<<dim3(H3_ / 32, (B_ * T_) / 256), 128, 0, stream>>>(xbf, k0T, b0, mx0);

  // Scan prologue: gru0 for t=0.
  gru0_step<<<dim3(H_ / 16, B_ / 16), 128, 0, stream>>>(
      mx0, rk0T, b0, len, h0bf[0], h0f, h0bf[1], 0);

  // Fused steps: K_t = { gru1(t), gru0(t+1) }, t = 0 .. T-2.
  for (int t = 0; t < T_ - 1; ++t) {
    gru_fused<<<dim3(H_ / 16, B_ / 16, 2), 128, 0, stream>>>(
        mx0 + (size_t)(t + 1) * B_ * H3_, rk0T, b0, k1T, rk1T, b1, len,
        /*h0cur=*/h0bf[(t + 1) & 1], /*h0nxt=*/h0bf[t & 1], h0f,
        /*h1cur=*/h1bf[t & 1], /*h1nxt=*/h1bf[(t + 1) & 1], h1f, out, t);
  }

  // Scan epilogue: gru1 for t = T-1.
  {
    const int t = T_ - 1;
    gru1_step<<<dim3(H_ / 16, B_ / 16), 128, 0, stream>>>(
        h0bf[(t + 1) & 1], h1bf[t & 1], k1T, rk1T, b1, len, h1f,
        h1bf[(t + 1) & 1], out, t);
  }
}
